// pool_60481729462503
// MI455X (gfx1250) — matrix operand order repaired and checked by reading, not compiled, not run on hardware
//
#include <hip/hip_runtime.h>

// ---------------------------------------------------------------------------
// CDNA5 (gfx1250) implementation of the CornerNet corner-pool block.
// Convs = implicit GEMMs on v_wmma_f32_16x16x32_bf16 (wave32).
// A-tile fed by TDM tensor_load_to_lds, B-tile by global_load_async_to_lds.
// ---------------------------------------------------------------------------

typedef __attribute__((ext_vector_type(16))) __bf16   bf16x16;
typedef __attribute__((ext_vector_type(8)))  __bf16   bf16x8;
typedef __attribute__((ext_vector_type(8)))  float    f32x8;
typedef __attribute__((ext_vector_type(4)))  unsigned u32x4;
typedef __attribute__((ext_vector_type(8)))  unsigned u32x8;

#define BDIM 4
#define HDIM 128
#define WDIM 128
#define NPIX (BDIM*HDIM*WDIM)          // 65536 pixels
#define KBLK 32
#define LROW 40                        // LDS row pitch in elems (64B + 16B pad)

static __device__ __forceinline__ bf16x8 zero8() {
    bf16x8 z;
#pragma unroll
    for (int i = 0; i < 8; ++i) z[i] = (__bf16)0.0f;
    return z;
}

// async 16B global -> LDS copy (per-lane, ASYNCcnt-tracked)
static __device__ __forceinline__ void async_b128(unsigned lds,
                                                  const __bf16* g) {
    asm volatile("global_load_async_to_lds_b128 %0, %1, off"
                 :: "v"(lds), "v"((unsigned long long)(uintptr_t)g)
                 : "memory");
}

// TDM 2-D tile load: 32 elems x 128 rows of bf16, row stride Ktot elems,
// LDS rows padded 64B data + 16B (pad_interval=16 DW, pad_amount=4 DW)
// so the DMA engine writes the bank-conflict-free LROW=40 pitch directly.
static __device__ __forceinline__ void tdm_load_A(const __bf16* Wt,
                                                  size_t elemOff,
                                                  unsigned ldsByte, int Ktot) {
    unsigned long long ga =
        (unsigned long long)(uintptr_t)Wt + (elemOff << 1);
    u32x4 g0;
    g0[0] = 1u;                                   // count=1, user mode
    g0[1] = ldsByte;                              // lds_addr
    g0[2] = (unsigned)ga;                         // global_addr[31:0]
    g0[3] = (unsigned)(ga >> 32) | (2u << 30);    // ga[56:32] | type=2
    u32x8 g1;
    g1[0] = (1u << 16)                            // data_size = 2B
          | (1u << 20)                            // pad_enable
          | (3u << 22)                            // pad_interval: 16 DWORDs
          | (3u << 25);                           // pad_amount : 4 DWORDs
    g1[1] = ((unsigned)Ktot & 0xFFFFu) << 16;     // tensor_dim0 lo16
    g1[2] = ((unsigned)Ktot >> 16) | (256u << 16);// tensor_dim0 hi | dim1 lo
    g1[3] = (32u << 16);                          // dim1 hi | tile_dim0 = 32
    g1[4] = 128u;                                 // tile_dim1=128, tile_dim2=0
    g1[5] = (unsigned)Ktot;                       // tensor_dim0_stride lo32
    g1[6] = 0u;                                   // stride0 hi | stride1 lo
    g1[7] = 0u;
    asm volatile("tensor_load_to_lds %0, %1" :: "s"(g0), "s"(g1) : "memory");
}

// ---------------------------------------------------------------------------
// fp32 NCHW -> bf16 NHWC
// ---------------------------------------------------------------------------
__global__ void cvt_nchw_to_nhwc(const float* __restrict__ in,
                                 __bf16* __restrict__ out) {
    size_t o = (size_t)blockIdx.x * blockDim.x + threadIdx.x;  // NHWC linear
    int c = (int)(o & 255);
    int w = (int)((o >> 8) & 127);
    int h = (int)((o >> 15) & 127);
    int n = (int)(o >> 22);
    out[o] = (__bf16)in[(((size_t)(n * 256 + c) * 128) + h) * 128 + w];
}

// ---------------------------------------------------------------------------
// fp32 OIHW -> bf16 [o][(kh*KW+kw)*I + i]   (GEMM-A layout, K contiguous)
// ---------------------------------------------------------------------------
__global__ void repack_w(const float* __restrict__ w, __bf16* __restrict__ out,
                         int I, int KHW) {
    size_t idx = (size_t)blockIdx.x * blockDim.x + threadIdx.x;  // over O*I*KHW
    int t   = (int)(idx % KHW);
    size_t r = idx / KHW;
    int i   = (int)(r % I);
    size_t o = r / I;
    out[o * (size_t)(I * KHW) + (size_t)t * I + i] = (__bf16)w[idx];
}

// ---------------------------------------------------------------------------
// Implicit-GEMM conv, bf16 WMMA, NHWC.
//   block: 256 thr (8 waves). tile: 128 outC x 128 pixels (one image row).
//   per wave: 32 outC x 64 pixels = 2x4 fragments of 16x16.
//   A tile via TDM (wave 0), B tile via per-lane async global->LDS.
// ---------------------------------------------------------------------------
template <int KW, int PAD>
__global__ __launch_bounds__(256)
void conv_wmma(const __bf16* __restrict__ X, const __bf16* __restrict__ Wt,
               __bf16* __restrict__ Out, int cinShift, int ldC, int ocBase0,
               int Ktot) {
    const int Cin    = 1 << cinShift;
    const int n      = blockIdx.x >> 7;
    const int h      = blockIdx.x & 127;
    const int ocBase = ocBase0 + blockIdx.y * 128;

    __shared__ __bf16 As[2][128 * LROW];
    __shared__ __bf16 Bs[2][128 * LROW];

    const int tid  = threadIdx.x;
    const int lane = tid & 31;
    const int wv   = tid >> 5;
    const int wvM  = wv & 3;   // outC sub-tile (32)
    const int wvN  = wv >> 2;  // pixel sub-tile (64)

    f32x8 acc[2][4];
#pragma unroll
    for (int mi = 0; mi < 2; ++mi)
#pragma unroll
        for (int ni = 0; ni < 4; ++ni)
#pragma unroll
            for (int e = 0; e < 8; ++e) acc[mi][ni][e] = 0.0f;

    const int nK = Ktot >> 5;

    // 512 16B B-chunks per tile (128 pixels x 4 parts); 2 per thread
    auto stageB = [&](int kb, int buf) {
        const int kstart = kb << 5;
        const int tap = kstart >> cinShift;
        const int icb = kstart & (Cin - 1);
        const int kh = tap / KW, kw = tap - kh * KW;
        const int ih = h + kh - PAD;
#pragma unroll
        for (int ci = 0; ci < 2; ++ci) {
            const int c   = tid + ci * 256;
            const int pix = c >> 2, part = c & 3;
            const int iw  = pix + kw - PAD;
            const int lo  = pix * LROW + part * 8;
            if (((unsigned)ih < 128u) & ((unsigned)iw < 128u)) {
                const __bf16* g = X + ((size_t)((n * 128 + ih) * 128 + iw) *
                                           Cin + icb + part * 8);
                async_b128((unsigned)(size_t)&Bs[buf][lo], g);
            } else {
                *(bf16x8*)&Bs[buf][lo] = zero8();   // halo -> zeros
            }
        }
    };
    auto stageA = [&](int kb, int buf) {
        tdm_load_A(Wt, (size_t)ocBase * Ktot + ((size_t)kb << 5),
                   (unsigned)(size_t)&As[buf][0], Ktot);
    };

    if (wv == 0) stageA(0, 0);
    stageB(0, 0);

    const int r  = lane & 15;
    const int hl = lane >> 4;

    for (int kb = 0; kb < nK; ++kb) {
        // publish tile kb: drain own counters, then block barrier
        if (wv == 0) asm volatile("s_wait_tensorcnt 0x0" ::: "memory");
        asm volatile("s_wait_asynccnt 0x0" ::: "memory");
        __syncthreads();
        const int cur = kb & 1;
        if (kb + 1 < nK) {          // overlap next tile DMA with the WMMAs
            if (wv == 0) stageA(kb + 1, cur ^ 1);
            stageB(kb + 1, cur ^ 1);
        }

        // A fragments: lane<16 holds K 0-7 & 16-23, lane>=16 holds 8-15 & 24-31
        bf16x16 af[2];
#pragma unroll
        for (int mi = 0; mi < 2; ++mi) {
            int base = (wvM * 32 + mi * 16 + r) * LROW + hl * 8;
            bf16x8 lo = *(const bf16x8*)&As[cur][base];
            bf16x8 hi = *(const bf16x8*)&As[cur][base + 16];
            af[mi] = __builtin_shufflevector(lo, hi, 0, 1, 2, 3, 4, 5, 6, 7, 8,
                                             9, 10, 11, 12, 13, 14, 15);
        }
        // B fragments: lane<16 = col holds K 0-7 & 16-23, lane>=16 holds 8-15 & 24-31
        bf16x16 bfr[4];
#pragma unroll
        for (int ni = 0; ni < 4; ++ni) {
            int base = (wvN * 64 + ni * 16 + r) * LROW + hl * 8;
            bf16x8 lo = *(const bf16x8*)&Bs[cur][base];
            bf16x8 hi = *(const bf16x8*)&Bs[cur][base + 16];
            bfr[ni] = __builtin_shufflevector(lo, hi, 0, 1, 2, 3, 4, 5, 6, 7, 8,
                                              9, 10, 11, 12, 13, 14, 15);
        }
#pragma unroll
        for (int mi = 0; mi < 2; ++mi)
#pragma unroll
            for (int ni = 0; ni < 4; ++ni)
                acc[mi][ni] = __builtin_amdgcn_wmma_f32_16x16x32_bf16(
                    false, af[mi], false, bfr[ni], (short)0, acc[mi][ni], false,
                    false);
    }

    // Epilogue: each lane owns 8 consecutive outC for one pixel per fragment.
    const size_t rowBase = (size_t)(n * 128 + h) * 128;
#pragma unroll
    for (int mi = 0; mi < 2; ++mi)
#pragma unroll
        for (int ni = 0; ni < 4; ++ni) {
            bf16x8 o;
#pragma unroll
            for (int e = 0; e < 8; ++e) o[e] = (__bf16)acc[mi][ni][e];
            int pix = wvN * 64 + ni * 16 + r;
            int oc  = ocBase + wvM * 32 + mi * 16 + hl * 8;
            *(bf16x8*)(Out + (rowBase + pix) * ldC + oc) = o;
        }
}

// ---------------------------------------------------------------------------
// BN stats: per-channel sum / sumsq over all pixels (bf16 NHWC input)
// ---------------------------------------------------------------------------
__global__ void bn_stats(const __bf16* __restrict__ x, float* __restrict__ sum,
                         float* __restrict__ sumsq, int C, int pixPerBlk) {
    int c = threadIdx.x;  // blockDim == C
    size_t p0 = (size_t)blockIdx.x * pixPerBlk;
    float s = 0.0f, q = 0.0f;
    for (int i = 0; i < pixPerBlk; ++i) {
        float v = (float)x[(p0 + i) * C + c];
        s += v;
        q += v * v;
    }
    atomicAdd(&sum[c], s);
    atomicAdd(&sumsq[c], q);
}

__global__ void bn_finalize(const float* __restrict__ sum,
                            const float* __restrict__ sumsq,
                            const float* __restrict__ g0,
                            const float* __restrict__ b0,
                            const float* __restrict__ g1,
                            const float* __restrict__ b1, int Csplit, int C,
                            float invN, float* __restrict__ scale,
                            float* __restrict__ shift) {
    int c = threadIdx.x;
    if (c >= C) return;
    float m = sum[c] * invN;
    float v = fmaxf(sumsq[c] * invN - m * m, 0.0f);
    float g = (c < Csplit) ? g0[c] : g1[c - Csplit];
    float b = (c < Csplit) ? b0[c] : b1[c - Csplit];
    float sc = g * rsqrtf(v + 1e-5f);
    scale[c] = sc;
    shift[c] = b - m * sc;
}

// ---------------------------------------------------------------------------
// Fused BN + ReLU + cummax along H (in place).  c < Csplit => reverse scan
// (branch-1 Top/Left pools), else forward (branch-2 Bottom/Right pools).
// thread = channel -> every scan step is a coalesced 512B line.
// ---------------------------------------------------------------------------
__global__ void bn_relu_cummax_h(__bf16* __restrict__ x,
                                 const float* __restrict__ scale,
                                 const float* __restrict__ shift, int C,
                                 int Csplit) {
    int c = threadIdx.x;
    int n = blockIdx.x >> 7, w = blockIdx.x & 127;
    bool rev = (c < Csplit);
    float sc = scale[c], sh = shift[c];
    float run = -1e30f;
    for (int i = 0; i < HDIM; ++i) {
        int h = rev ? (HDIM - 1 - i) : i;
        size_t off = ((size_t)(n * 128 + h) * 128 + w) * C + c;
        float v = fmaxf((float)x[off] * sc + sh, 0.0f);
        run = fmaxf(run, v);
        x[off] = (__bf16)run;
    }
}

__global__ void cummax_w(__bf16* __restrict__ x, int C, int Csplit) {
    int c = threadIdx.x;
    int n = blockIdx.x >> 7, h = blockIdx.x & 127;
    bool rev = (c < Csplit);
    float run = -1e30f;
    for (int i = 0; i < WDIM; ++i) {
        int w = rev ? (WDIM - 1 - i) : i;
        size_t off = ((size_t)(n * 128 + h) * 128 + w) * C + c;
        run = fmaxf(run, (float)x[off]);
        x[off] = (__bf16)run;
    }
}

// BN + ReLU elementwise in place (bf16 NHWC)
__global__ void bn_relu_ip(__bf16* __restrict__ x,
                           const float* __restrict__ scale,
                           const float* __restrict__ shift, int Cmask) {
    size_t i = (size_t)blockIdx.x * blockDim.x + threadIdx.x;
    int c = (int)(i & Cmask);
    float v = (float)x[i] * scale[c] + shift[c];
    x[i] = (__bf16)fmaxf(v, 0.0f);
}

// BN + ReLU + NHWC(bf16) -> NCHW(fp32) output
__global__ void bn_relu_to_nchw(const __bf16* __restrict__ x,
                                const float* __restrict__ scale,
                                const float* __restrict__ shift,
                                float* __restrict__ out) {
    size_t i = (size_t)blockIdx.x * blockDim.x + threadIdx.x;  // NCHW linear
    int w = (int)(i & 127);
    int h = (int)((i >> 7) & 127);
    int c = (int)((i >> 14) & 255);
    int n = (int)(i >> 22);
    float v = (float)x[((size_t)(n * 128 + h) * 128 + w) * 256 + c];
    out[i] = fmaxf(v * scale[c] + shift[c], 0.0f);
}

// ---------------------------------------------------------------------------
// Host orchestration
// ---------------------------------------------------------------------------
extern "C" void kernel_launch(void* const* d_in, const int* in_sizes, int n_in,
                              void* d_out, int out_size, void* d_ws,
                              size_t ws_size, hipStream_t stream) {
    const float* x    = (const float*)d_in[0];
    const float* w_p1 = (const float*)d_in[1];
    const float* g_p1 = (const float*)d_in[2];
    const float* b_p1 = (const float*)d_in[3];
    const float* w_p2 = (const float*)d_in[4];
    const float* g_p2 = (const float*)d_in[5];
    const float* b_p2 = (const float*)d_in[6];
    const float* w_p  = (const float*)d_in[7];
    const float* g_p  = (const float*)d_in[8];
    const float* b_p  = (const float*)d_in[9];
    const float* w_c1 = (const float*)d_in[10];
    const float* g_c1 = (const float*)d_in[11];
    const float* b_c1 = (const float*)d_in[12];
    const float* w_c2 = (const float*)d_in[13];
    const float* g_c2 = (const float*)d_in[14];
    const float* b_c2 = (const float*)d_in[15];
    float* out = (float*)d_out;

    // workspace carve-up (256B aligned blocks)
    uintptr_t base = (uintptr_t)d_ws;
    auto alloc = [&](size_t bytes) -> void* {
        void* p = (void*)base;
        base += (bytes + 255) & ~(size_t)255;
        return p;
    };
    __bf16* xh     = (__bf16*)alloc((size_t)NPIX * 256 * 2);  // x bf16 NHWC
    __bf16* wpk    = (__bf16*)alloc((size_t)256 * 2304 * 2);  // w_p1|w_p2
    __bf16* wp     = (__bf16*)alloc((size_t)256 * 2304 * 2);
    __bf16* wc1    = (__bf16*)alloc((size_t)256 * 256 * 2);
    __bf16* wc2    = (__bf16*)alloc((size_t)256 * 4608 * 2);
    __bf16* conv12 = (__bf16*)alloc((size_t)NPIX * 256 * 2);  // branches
    __bf16* relu1  = (__bf16*)alloc((size_t)NPIX * 512 * 2);  // concat feats
    float* sum     = (float*)alloc(512 * 4);
    float* sumsq   = (float*)alloc(512 * 4);
    float* scale   = (float*)alloc(512 * 4);
    float* shift   = (float*)alloc(512 * 4);
    __bf16* finraw = xh;  // reuse x buffer after its last consumer (conv c1)

    const float invN = 1.0f / (float)NPIX;
    const dim3 cgrid(BDIM * HDIM, 2);  // 512 pixel-rows x 2 outC tiles

    // 0) layout conversions
    cvt_nchw_to_nhwc<<<NPIX * 256 / 256, 256, 0, stream>>>(x, xh);
    repack_w<<<(128 * 256 * 9) / 256, 256, 0, stream>>>(w_p1, wpk, 256, 9);
    repack_w<<<(128 * 256 * 9) / 256, 256, 0, stream>>>(
        w_p2, wpk + (size_t)128 * 2304, 256, 9);
    repack_w<<<(256 * 256 * 9) / 256, 256, 0, stream>>>(w_p, wp, 256, 9);
    repack_w<<<(256 * 256 * 1) / 256, 256, 0, stream>>>(w_c1, wc1, 256, 1);
    repack_w<<<(256 * 512 * 9) / 256, 256, 0, stream>>>(w_c2, wc2, 512, 9);

    // 1) fused branch convs: 256 outC (p1 = ch 0-127, p2 = ch 128-255)
    conv_wmma<3, 1><<<cgrid, 256, 0, stream>>>(xh, wpk, conv12, 8, 256, 0, 2304);
    hipMemsetAsync(sum, 0, 2 * 512 * 4, stream);
    bn_stats<<<256, 256, 0, stream>>>(conv12, sum, sumsq, 256, 256);
    bn_finalize<<<1, 512, 0, stream>>>(sum, sumsq, g_p1, b_p1, g_p2, b_p2, 128,
                                       256, invN, scale, shift);
    // 2) BN+ReLU+corner pools in place (rev for branch1, fwd for branch2)
    bn_relu_cummax_h<<<BDIM * WDIM, 256, 0, stream>>>(conv12, scale, shift, 256,
                                                      128);
    cummax_w<<<BDIM * HDIM, 256, 0, stream>>>(conv12, 256, 128);

    // 3) merge conv (-> relu1 ch 0-255) and skip 1x1 conv (-> ch 256-511)
    conv_wmma<3, 1><<<cgrid, 256, 0, stream>>>(conv12, wp, relu1, 8, 512, 0,
                                               2304);
    conv_wmma<1, 0><<<cgrid, 256, 0, stream>>>(xh, wc1, relu1, 8, 512, 256, 256);
    hipMemsetAsync(sum, 0, 2 * 512 * 4, stream);
    bn_stats<<<128, 512, 0, stream>>>(relu1, sum, sumsq, 512, 512);
    bn_finalize<<<1, 512, 0, stream>>>(sum, sumsq, g_p, b_p, g_c1, b_c1, 256,
                                       512, invN, scale, shift);
    bn_relu_ip<<<(NPIX * 512) / 256, 256, 0, stream>>>(relu1, scale, shift,
                                                       511);

    // 4) final conv (512 -> 256), BN, ReLU, NCHW fp32 out
    conv_wmma<3, 1><<<cgrid, 256, 0, stream>>>(relu1, wc2, finraw, 9, 256, 0,
                                               4608);
    hipMemsetAsync(sum, 0, 2 * 512 * 4, stream);
    bn_stats<<<256, 256, 0, stream>>>(finraw, sum, sumsq, 256, 256);
    bn_finalize<<<1, 512, 0, stream>>>(sum, sumsq, g_c2, b_c2, g_c2, b_c2, 256,
                                       256, invN, scale, shift);
    bn_relu_to_nchw<<<(NPIX * 256) / 256, 256, 0, stream>>>(finraw, scale,
                                                            shift, out);
}
